// ROSA1bitLayer_52295521796232
// MI455X (gfx1250) — matrix operand order repaired and checked by reading, not compiled, not run on hardware
//
#include <hip/hip_runtime.h>
#include <hip/hip_bf16.h>
#include <stdint.h>

#define Bn 4
#define Tn 1024
#define Cn 4096
#define NS 32
#define EMBn 64
#define LN_EPS 1e-5f
#define SMAX (2 * Tn + 1)   // 2049 suffix-automaton nodes max

typedef __bf16 v16bf __attribute__((ext_vector_type(16)));
typedef __bf16 v8bf  __attribute__((ext_vector_type(8)));
typedef float  v8f   __attribute__((ext_vector_type(8)));
typedef unsigned int u32x4 __attribute__((ext_vector_type(4)));
typedef int          i32x4 __attribute__((ext_vector_type(4)));
typedef int          i32x8 __attribute__((ext_vector_type(8)));

// ---------------------------------------------------------------------------
// Prep: convert w_bits [32][4096] and w_out [4096][64] to bf16 once so the
// GEMM kernels load B fragments directly (no per-block re-conversion).
// ---------------------------------------------------------------------------
__global__ __launch_bounds__(256) void cvt_weights_kernel(
    const float* __restrict__ w_bits, const float* __restrict__ w_out,
    unsigned short* __restrict__ wb_bf, unsigned short* __restrict__ wo_bf) {
    const int i = blockIdx.x * 256 + threadIdx.x;
    if (i < NS * Cn)   ((__bf16*)wb_bf)[i] = (__bf16)w_bits[i];
    if (i < Cn * EMBn) ((__bf16*)wo_bf)[i] = (__bf16)w_out[i];
}

// ---------------------------------------------------------------------------
// Kernel A: logits = x @ w_bits^T, binarize, write per-stream bit sequences.
// One block per 16-row M tile; 4 waves split K (1024 each) for 4x the memory-
// level parallelism; both 16-wide N tiles (NS=32) stay in the same wave so x
// is streamed from HBM exactly once. Partial accumulators reduce through LDS.
// ---------------------------------------------------------------------------
__global__ __launch_bounds__(128) void rosa_bits_kernel(
    const float* __restrict__ x, const unsigned short* __restrict__ wb_bf_u16,
    unsigned char* __restrict__ bitbuf) {
    __shared__ float pacc[4][32][16];
    const __bf16* wb_bf = (const __bf16*)wb_bf_u16;
    const int lane = threadIdx.x & 31;
    const int wave = threadIdx.x >> 5;    // 0..3 : K splitter
    const int mt   = blockIdx.x;          // 0..255
    const int hf   = lane >> 4;
    const int l16  = lane & 15;

    const float*  xrow = x + (size_t)(mt * 16 + l16) * Cn;    // A row M = l16
    const __bf16* wb0  = wb_bf + (size_t)l16 * Cn;            // stream l16
    const __bf16* wb1  = wb_bf + (size_t)(l16 + 16) * Cn;     // stream l16+16
    const int ako = hf * 16;   // A K-chunk base within 32-wide K tile
    const int bko = hf * 16;   // B K base within 32-wide K tile

    v8f acc0 = {}; v8f acc1 = {};
    const int kbeg = wave * (Cn / 4);
    for (int kk = kbeg; kk < kbeg + Cn / 4; kk += 128) {
        // speculative prefetch of the x stream ~4KB ahead (global_prefetch_b8)
        __builtin_prefetch((const void*)(xrow + kk + 1024 + ako * 8), 0, 1);
#pragma unroll
        for (int ks = 0; ks < 128; ks += 32) {
            const int k0 = kk + ks;
            float4 a0f = *(const float4*)(xrow + k0 + ako);
            float4 a1f = *(const float4*)(xrow + k0 + ako + 4);
            float4 a2f = *(const float4*)(xrow + k0 + ako + 8);
            float4 a3f = *(const float4*)(xrow + k0 + ako + 12);
            v16bf a;
            a[0]  = (__bf16)a0f.x; a[1]  = (__bf16)a0f.y; a[2]  = (__bf16)a0f.z; a[3]  = (__bf16)a0f.w;
            a[4]  = (__bf16)a1f.x; a[5]  = (__bf16)a1f.y; a[6]  = (__bf16)a1f.z; a[7]  = (__bf16)a1f.w;
            a[8]  = (__bf16)a2f.x; a[9]  = (__bf16)a2f.y; a[10] = (__bf16)a2f.z; a[11] = (__bf16)a2f.w;
            a[12] = (__bf16)a3f.x; a[13] = (__bf16)a3f.y; a[14] = (__bf16)a3f.z; a[15] = (__bf16)a3f.w;

            v16bf b0 = *(const v16bf*)(wb0 + k0 + bko);
            v16bf b1 = *(const v16bf*)(wb1 + k0 + bko);

            acc0 = __builtin_amdgcn_wmma_f32_16x16x32_bf16(false, a, false, b0,
                                                           (short)0, acc0, false, false);
            acc1 = __builtin_amdgcn_wmma_f32_16x16x32_bf16(false, a, false, b1,
                                                           (short)0, acc1, false, false);
        }
    }

#pragma unroll
    for (int j = 0; j < 8; ++j) {
        pacc[wave][lane][j]     = acc0[j];
        pacc[wave][lane][8 + j] = acc1[j];
    }
    __syncthreads();

    if (wave == 0) {
        v8f s0 = {}; v8f s1 = {};
#pragma unroll
        for (int w = 0; w < 4; ++w)
#pragma unroll
            for (int j = 0; j < 8; ++j) {
                s0[j] += pacc[w][lane][j];
                s1[j] += pacc[w][lane][8 + j];
            }
        // D layout: lane holds N = l16, rows M = r + 8*hf.  bit = logit > 0.
        for (int r = 0; r < 8; ++r) {
            int row = mt * 16 + r + hf * 8;          // global b*T + t
            int bb = row >> 10, tt = row & (Tn - 1);
            bitbuf[(size_t)(bb * NS + l16) * Tn + tt]      = s0[r] > 0.f ? 1 : 0;
            bitbuf[(size_t)(bb * NS + l16 + 16) * Tn + tt] = s1[r] > 0.f ? 1 : 0;
        }
    }
}

// ---------------------------------------------------------------------------
// Kernel B: online 1-bit suffix-automaton (ROSA) — one workgroup per stream.
// Node arrays live entirely in LDS (~42 KB << 320 KB/WGP). The bit sequence
// is staged into LDS via the Tensor Data Mover (1024B 1-row tile, D# built
// per CDNA5 ISA), synchronized with s_wait_tensorcnt.
// The construction itself is inherently sequential -> lane 0 runs it.
// ---------------------------------------------------------------------------
__global__ __launch_bounds__(32) void rosa_automaton_kernel(
    const unsigned char* __restrict__ bitbuf, unsigned char* __restrict__ tokbuf) {
    __shared__ int trans[SMAX * 2];
    __shared__ int slink[SMAX];
    __shared__ int sd[SMAX];
    __shared__ int se[SMAX];
    __shared__ unsigned char xb[Tn];

    const int sid  = blockIdx.x;   // 0..127 == b*NS + s
    const int lane = threadIdx.x;

    for (int i = lane; i < SMAX; i += 32) {
        trans[2 * i] = -1; trans[2 * i + 1] = -1;
        slink[i] = -1; sd[i] = 0; se[i] = -1;
    }

    // --- TDM: async-load 1024 bytes bitbuf[sid*Tn .. +1023] into xb[] ---
    {
        const uint64_t gaddr = (uint64_t)(uintptr_t)(bitbuf + (size_t)sid * Tn);
        const uint32_t laddr = (uint32_t)(uintptr_t)(&xb[0]);   // LDS byte offset
        u32x4 g0;
        g0[0] = 1u;                                   // count=1, user mode
        g0[1] = laddr;                                // lds_addr
        g0[2] = (uint32_t)gaddr;                      // global_addr[31:0]
        g0[3] = (uint32_t)((gaddr >> 32) & 0x1FFFFFFu) | (2u << 30); // [56:32] | type=2
        i32x8 g1;
        g1[0] = 0;                                    // wg_mask=0 (not in cluster), 1B elems
        g1[1] = (int)(1024u << 16);                   // tensor_dim0[15:0]=1024
        g1[2] = (int)(1u << 16);                      // tensor_dim0 hi=0 | tensor_dim1=1
        g1[3] = (int)(1024u << 16);                   // tensor_dim1 hi=0 | tile_dim0=1024
        g1[4] = 1;                                    // tile_dim1=1, tile_dim2=0
        g1[5] = 1024;                                 // tensor_dim0_stride lo
        g1[6] = 0; g1[7] = 0;                         // stride hi, dim1_stride
        i32x4 g2 = {0, 0, 0, 0};
        i32x4 g3 = {0, 0, 0, 0};
#if defined(__clang_major__) && (__clang_major__ >= 23)
        i32x8 gx = {0, 0, 0, 0, 0, 0, 0, 0};
        __builtin_amdgcn_tensor_load_to_lds(g0, g1, g2, g3, gx, 0);
#else
        __builtin_amdgcn_tensor_load_to_lds(g0, g1, g2, g3, 0);
#endif
        __builtin_amdgcn_s_wait_tensorcnt(0);
    }
    __syncthreads();

    if (lane == 0) {
        const int bb = sid / NS, ss = sid % NS;
        unsigned char* tout = tokbuf + (size_t)bb * Tn * NS + ss;
        int g = 0, z = 1;
        for (int i = 0; i < Tn; ++i) {
            const int t = xb[i];
            int r = z++; sd[r] = sd[g] + 1;
            int p = g;
            while (p != -1 && trans[2 * p + t] == -1) { trans[2 * p + t] = r; p = slink[p]; }
            if (p == -1) {
                slink[r] = 0;
            } else {
                int q = trans[2 * p + t];
                if (sd[p] + 1 == sd[q]) {
                    slink[r] = q;
                } else {
                    int u = z++;
                    trans[2 * u] = trans[2 * q]; trans[2 * u + 1] = trans[2 * q + 1];
                    sd[u] = sd[p] + 1;
                    slink[u] = slink[q];
                    se[u] = se[q];
                    while (p != -1 && trans[2 * p + t] == q) { trans[2 * p + t] = u; p = slink[p]; }
                    slink[q] = u; slink[r] = u;
                }
            }
            // query: first v on suffix chain from OLD last with d[v]>0 && e[v]>=0
            int v = g;
            while (v != -1 && !(sd[v] > 0 && se[v] >= 0)) v = slink[v];
            int a;
            if (v == -1) a = 2;
            else { int nxt = se[v] + 1; a = (nxt < Tn) ? (int)xb[nxt] : 2; }
            tout[(size_t)i * NS] = (unsigned char)a;
            // update end positions along chain from OLD last
            int vv = g;
            while (vv != -1 && se[vv] < i) { se[vv] = i; vv = slink[vv]; }
            g = r;
        }
    }
}

// ---------------------------------------------------------------------------
// Kernel B2: token counts -> embedding blend -> pooled[4096][64] in bf16.
// ---------------------------------------------------------------------------
__global__ __launch_bounds__(256) void pool_kernel(
    const unsigned char* __restrict__ tokbuf, const float* __restrict__ emb_table,
    unsigned short* __restrict__ pooled_u16) {
    __shared__ float semb[3 * EMBn];
    for (int i = threadIdx.x; i < 3 * EMBn; i += 256) semb[i] = emb_table[i];
    __syncthreads();

    const int bt = blockIdx.x * 256 + threadIdx.x;   // 0..4095
    const unsigned char* tp = tokbuf + (size_t)bt * NS;
    int c0 = 0, c1 = 0, c2 = 0;
    for (int s = 0; s < NS; ++s) {
        int v = tp[s];
        c0 += (v == 0); c1 += (v == 1); c2 += (v == 2);
    }
    const float f0 = c0 * (1.f / NS), f1 = c1 * (1.f / NS), f2 = c2 * (1.f / NS);
    __bf16* pout = (__bf16*)(pooled_u16 + (size_t)bt * EMBn);
    for (int e = 0; e < EMBn; ++e) {
        float v = f0 * semb[e] + f1 * semb[EMBn + e] + f2 * semb[2 * EMBn + e];
        pout[e] = (__bf16)v;
    }
}

// ---------------------------------------------------------------------------
// Kernel C: out = pooled @ w_out^T via WMMA (K=64 -> 2 bf16 WMMAs per tile).
// One block per 16-row M tile; 8 waves sweep the 256 N tiles; the A fragment
// is loaded once per wave and reused across the entire N sweep.
// ---------------------------------------------------------------------------
__global__ __launch_bounds__(256) void out_gemm_kernel(
    const unsigned short* __restrict__ pooled_u16,
    const unsigned short* __restrict__ wo_bf_u16, float* __restrict__ out) {
    const __bf16* pooled = (const __bf16*)pooled_u16;
    const __bf16* wo_bf  = (const __bf16*)wo_bf_u16;
    const int mt   = blockIdx.x;
    const int lane = threadIdx.x & 31;
    const int wave = threadIdx.x >> 5;
    const int hf   = lane >> 4, l16 = lane & 15;

    // A fragments for k0=0 and k0=32
    const __bf16* prow = pooled + (size_t)(mt * 16 + l16) * EMBn;
    v8bf p0 = *(const v8bf*)(prow + hf * 16);
    v8bf p1 = *(const v8bf*)(prow + hf * 16 + 8);
    v8bf p2 = *(const v8bf*)(prow + 32 + hf * 16);
    v8bf p3 = *(const v8bf*)(prow + 32 + hf * 16 + 8);
    v16bf a0, a1;
#pragma unroll
    for (int j = 0; j < 8; ++j) {
        a0[j] = p0[j]; a0[j + 8] = p1[j];
        a1[j] = p2[j]; a1[j + 8] = p3[j];
    }

    for (int nt = wave; nt < Cn / 16; nt += 8) {
        const int cc = nt * 16 + l16;                       // output column N
        const __bf16* wrow = wo_bf + (size_t)cc * EMBn;     // B[k][cc] = w_out[cc][k]
        v16bf b0 = *(const v16bf*)(wrow + hf * 16);
        v16bf b1 = *(const v16bf*)(wrow + 32 + hf * 16);
        v8f acc = {};
        acc = __builtin_amdgcn_wmma_f32_16x16x32_bf16(false, a0, false, b0,
                                                      (short)0, acc, false, false);
        acc = __builtin_amdgcn_wmma_f32_16x16x32_bf16(false, a1, false, b1,
                                                      (short)0, acc, false, false);
#pragma unroll
        for (int r = 0; r < 8; ++r) {
            int m = r + hf * 8;
            out[(size_t)(mt * 16 + m) * Cn + cc] = acc[r];
        }
    }
}

// ---------------------------------------------------------------------------
// Kernel D: LayerNorm in place over each row of out (C=4096).
// ---------------------------------------------------------------------------
__global__ __launch_bounds__(256) void ln_kernel(
    float* __restrict__ out, const float* __restrict__ ln_w,
    const float* __restrict__ ln_b) {
    const int row = blockIdx.x;
    float* orow = out + (size_t)row * Cn;
    float s = 0.f, sq = 0.f;
    for (int c = threadIdx.x; c < Cn; c += 256) {
        float v = orow[c]; s += v; sq += v * v;
    }
    for (int off = 16; off > 0; off >>= 1) {
        s  += __shfl_xor(s, off, 32);
        sq += __shfl_xor(sq, off, 32);
    }
    __shared__ float rs[8], rq[8];
    const int wave = threadIdx.x >> 5, lane = threadIdx.x & 31;
    if (lane == 0) { rs[wave] = s; rq[wave] = sq; }
    __syncthreads();
    if (threadIdx.x == 0) {
        float ts = 0.f, tq = 0.f;
        for (int w = 0; w < 8; ++w) { ts += rs[w]; tq += rq[w]; }
        rs[0] = ts; rq[0] = tq;
    }
    __syncthreads();
    const float mean = rs[0] * (1.f / Cn);
    const float var  = rq[0] * (1.f / Cn) - mean * mean;
    const float rstd = rsqrtf(var + LN_EPS);
    for (int c = threadIdx.x; c < Cn; c += 256) {
        orow[c] = (orow[c] - mean) * rstd * ln_w[c] + ln_b[c];
    }
}

extern "C" void kernel_launch(void* const* d_in, const int* in_sizes, int n_in,
                              void* d_out, int out_size, void* d_ws, size_t ws_size,
                              hipStream_t stream) {
    const float* x         = (const float*)d_in[0];
    const float* w_bits    = (const float*)d_in[1];
    const float* emb_table = (const float*)d_in[2];
    const float* w_out     = (const float*)d_in[3];
    const float* ln_w      = (const float*)d_in[4];
    const float* ln_b      = (const float*)d_in[5];
    float* out = (float*)d_out;

    // scratch: bits(128K) | tokens(128K) | pooled bf16(512K) | wb bf16(256K) | wo bf16(512K)
    unsigned char*  bitbuf = (unsigned char*)d_ws;
    unsigned char*  tokbuf = bitbuf + (size_t)Bn * NS * Tn;
    unsigned short* pooled = (unsigned short*)(tokbuf + (size_t)Bn * Tn * NS);
    unsigned short* wb_bf  = pooled + (size_t)Bn * Tn * EMBn;
    unsigned short* wo_bf  = wb_bf + (size_t)NS * Cn;

    cvt_weights_kernel<<<(Cn * EMBn) / 256, 256, 0, stream>>>(w_bits, w_out, wb_bf, wo_bf);
    rosa_bits_kernel<<<(Bn * Tn) / 16, 128, 0, stream>>>(x, wb_bf, bitbuf);
    rosa_automaton_kernel<<<Bn * NS, 32, 0, stream>>>(bitbuf, tokbuf);
    pool_kernel<<<(Bn * Tn) / 256, 256, 0, stream>>>(tokbuf, emb_table, pooled);
    out_gemm_kernel<<<(Bn * Tn) / 16, 256, 0, stream>>>(pooled, wo_bf, out);
    ln_kernel<<<Bn * Tn, 256, 0, stream>>>(out, ln_w, ln_b);
}
